// MultiHeadAttention_81329500717159
// MI455X (gfx1250) — matrix operand order repaired and checked by reading, not compiled, not run on hardware
//
#include <hip/hip_runtime.h>

// ---------------- problem constants ----------------
#define B_   2
#define T_   2048
#define D_   2048
#define H_   16
#define DH_  128
#define BT_  (B_ * T_)

typedef __attribute__((ext_vector_type(16))) _Float16 v16h;
typedef __attribute__((ext_vector_type(8)))  _Float16 v8h;
typedef __attribute__((ext_vector_type(8)))  float    v8f;
typedef __attribute__((ext_vector_type(4)))  int      v4i;

#define NEG_HUGE (-3.0e38f)

#if __has_builtin(__builtin_amdgcn_global_load_async_to_lds_b128)
#define HAVE_ASYNC_LDS 1
#else
#define HAVE_ASYNC_LDS 0
#endif

static __device__ __forceinline__ v8f wmma_f16(v16h a, v16h b, v8f c) {
  // D = A(16x32 f16) * B(32x16 f16) + C(16x16 f32)
  return __builtin_amdgcn_wmma_f32_16x16x32_f16(
      /*neg_a=*/false, a, /*neg_b=*/false, b,
      /*c_mod=*/(short)0, c, /*reuse_a=*/false, /*reuse_b=*/false);
}

// A operand, 16x32 f16, from row-major memory (element (m,k) at m*lda + k).
// Layout: lanes 0-15 -> M=lane, K 0-15; lanes 16-31 -> K 16-31 (contiguous, matches ldBt).
static __device__ __forceinline__ v16h ldA16x32(const _Float16* __restrict__ base, int lda) {
  const int lane = threadIdx.x & 31;
  const int m  = lane & 15;
  const int kb = (lane < 16) ? 0 : 16;
  const _Float16* p = base + (long)m * lda + kb;
  v8h lo = *(const v8h*)(p);        // K = kb .. kb+7
  v8h hi = *(const v8h*)(p + 8);    // K = kb+8 .. kb+15
  v16h a;
#pragma unroll
  for (int i = 0; i < 8; ++i) { a[i] = lo[i]; a[i + 8] = hi[i]; }
  return a;
}

// B operand, 32x16 f16, where memory holds the TRANSPOSE row-major:
// element (k,n) at n*ld + k  -> per-lane contiguous 32B load (global or LDS).
static __device__ __forceinline__ v16h ldBt(const _Float16* __restrict__ base, int ld) {
  const int lane = threadIdx.x & 31;
  const int n  = lane & 15;
  const int kb = (lane < 16) ? 0 : 16;
  return *(const v16h*)(base + (long)n * ld + kb);
}

// 16-byte global -> LDS copy chunk (async if available on this toolchain)
static __device__ __forceinline__ void cp16(const _Float16* g, _Float16* l) {
#if HAVE_ASYNC_LDS
  _Float16* gnc = const_cast<_Float16*>(g);
  __builtin_amdgcn_global_load_async_to_lds_b128(
      (__attribute__((address_space(1))) v4i*)gnc,
      (__attribute__((address_space(3))) v4i*)l,
      0, 0);
#else
  *(v8h*)l = *(const v8h*)g;
#endif
}

static __device__ __forceinline__ void wait_async0() {
#if __has_builtin(__builtin_amdgcn_s_wait_asynccnt)
  __builtin_amdgcn_s_wait_asynccnt(0);
#elif HAVE_ASYNC_LDS
  asm volatile("s_wait_asynccnt 0" ::: "memory");
#endif
}

// ---------------- conversion / transpose kernels ----------------

__global__ void k_cvt_f32_f16(const float* __restrict__ in, _Float16* __restrict__ out, int n) {
  int i = blockIdx.x * blockDim.x + threadIdx.x;
  if (i < n) out[i] = (_Float16)in[i];
}

// in: [H][D][Dh] f32  ->  out: [H][Dh][D] f16 (B-operand loads become contiguous)
__global__ void k_cvt_whdt(const float* __restrict__ in, _Float16* __restrict__ out) {
  int i = blockIdx.x * blockDim.x + threadIdx.x;
  if (i >= H_ * D_ * DH_) return;
  int h = i / (D_ * DH_);
  int r = i - h * (D_ * DH_);
  int d = r / DH_;
  int e = r - d * DH_;
  out[(long)h * DH_ * D_ + (long)e * D_ + d] = (_Float16)in[i];
}

// in: Wp [D][D] f32 -> out: WpT [D][D] f16 with out[e][d] = in[d][e]
__global__ void k_cvt_wpt(const float* __restrict__ in, _Float16* __restrict__ out) {
  int i = blockIdx.x * blockDim.x + threadIdx.x;
  if (i >= D_ * D_) return;
  int d = i / D_;
  int e = i - d * D_;
  out[(long)e * D_ + d] = (_Float16)in[i];
}

// ---------------- QKV projection: one wave computes 32 x 128 ----------------
// grid: (BT/32, 3*H), block: 32.  V is stored TRANSPOSED: Vt[(bh*Dh+e)*T + t].
__global__ void __launch_bounds__(32, 1)
k_qkv(const _Float16* __restrict__ xh,
      const _Float16* __restrict__ WqT, const _Float16* __restrict__ WkT,
      const _Float16* __restrict__ WvT,
      const float* __restrict__ bq, const float* __restrict__ bk,
      const float* __restrict__ bv,
      _Float16* __restrict__ Q, _Float16* __restrict__ K,
      _Float16* __restrict__ Vt) {
  const int mb  = blockIdx.x;
  const int sel = blockIdx.y / H_;   // 0=Q 1=K 2=V
  const int h   = blockIdx.y % H_;
  const _Float16* WT  = (sel == 0) ? WqT : (sel == 1) ? WkT : WvT;
  const float*   bias = (sel == 0) ? bq  : (sel == 1) ? bk  : bv;

  const int row0 = mb * 32;            // row in flattened [B*T]
  const int b    = row0 / T_;
  const int t0   = row0 % T_;
  const int bh   = b * H_ + h;
  const int lane = threadIdx.x & 31;
  const int n    = lane & 15;
  const int mrow = (lane >= 16) ? 8 : 0;

  v8f zero = {0.f,0.f,0.f,0.f,0.f,0.f,0.f,0.f};
  v8f acc[2][8];
#pragma unroll
  for (int mt = 0; mt < 2; ++mt)
#pragma unroll
    for (int i = 0; i < 8; ++i) acc[mt][i] = zero;

  const _Float16* Wh = WT + (long)h * DH_ * D_;
  for (int k0 = 0; k0 < D_; k0 += 32) {
    v16h a0 = ldA16x32(xh + (long)row0 * D_ + k0, D_);
    v16h a1 = ldA16x32(xh + (long)(row0 + 16) * D_ + k0, D_);
#pragma unroll
    for (int g = 0; g < 2; ++g) {
      v16h bm[4];
#pragma unroll
      for (int t = 0; t < 4; ++t)
        bm[t] = ldBt(Wh + (long)((g * 4 + t) * 16) * D_ + k0, D_);
#pragma unroll
      for (int t = 0; t < 4; ++t) {
        acc[0][g * 4 + t] = wmma_f16(a0, bm[t], acc[0][g * 4 + t]);
        acc[1][g * 4 + t] = wmma_f16(a1, bm[t], acc[1][g * 4 + t]);
      }
    }
  }

#pragma unroll
  for (int mt = 0; mt < 2; ++mt) {
#pragma unroll
    for (int nt = 0; nt < 8; ++nt) {
#pragma unroll
      for (int j = 0; j < 8; ++j) {
        int e = nt * 16 + n;
        int t = t0 + mt * 16 + j + mrow;
        float v = acc[mt][nt][j] + bias[h * DH_ + e];
        if (sel == 2)
          Vt[((long)bh * DH_ + e) * T_ + t] = (_Float16)v;       // transposed
        else if (sel == 0)
          Q[((long)bh * T_ + t) * DH_ + e] = (_Float16)v;
        else
          K[((long)bh * T_ + t) * DH_ + e] = (_Float16)v;
      }
    }
  }
}

// ---------------- flash attention: one wave per (b,h, 16 query rows) ----------------
// grid: (T/16, B*H), block: 32.  K and V tiles double-buffered in LDS via async copies.
__global__ void __launch_bounds__(32, 1)
k_flash(const _Float16* __restrict__ Q, const _Float16* __restrict__ Kg,
        const _Float16* __restrict__ Vt, _Float16* __restrict__ O) {
  const int qb  = blockIdx.x;
  const int bh  = blockIdx.y;
  const int b   = bh / H_;
  const int h   = bh % H_;
  const int q0  = qb * 16;
  const int lane = threadIdx.x & 31;
  const int n    = lane & 15;
  const int mrow = (lane >= 16) ? 8 : 0;

  __shared__ __attribute__((aligned(16))) _Float16 Ksh[2][32 * 128];  // [key][d], 8KB each
  __shared__ __attribute__((aligned(16))) _Float16 Vsh[2][128 * 32];  // [e][key], 8KB each
  __shared__ __attribute__((aligned(16))) _Float16 Psh[16][40];       // padded rows

  v8f zero = {0.f,0.f,0.f,0.f,0.f,0.f,0.f,0.f};
  v8f o[8];
  float m[8], l[8];
#pragma unroll
  for (int i = 0; i < 8; ++i) { o[i] = zero; m[i] = NEG_HUGE; l[i] = 0.f; }

  // Q tile 16 x 128 in registers as 4 A-operands
  const _Float16* Qbase = Q + ((long)bh * T_ + q0) * DH_;
  v16h qa[4];
#pragma unroll
  for (int d = 0; d < 4; ++d) qa[d] = ldA16x32(Qbase + d * 32, DH_);

  const float scale = 0.08838834764831845f;  // 1/sqrt(128)
  const int nkb = (q0 + 15) / 32 + 1;        // causal: 32-wide key blocks

  const _Float16* KgBH = Kg + (long)bh * T_ * DH_;
  const _Float16* VtBH = Vt + (long)bh * DH_ * T_;

  // stage K tile (8192 contiguous bytes) into Ksh[buf]
  auto stageK = [&](int buf, int key0) {
    const _Float16* g = KgBH + (long)key0 * DH_;
    _Float16* dst = &Ksh[buf][0];
#pragma unroll
    for (int j = 0; j < 16; ++j)
      cp16(g + lane * 8 + j * 256, dst + lane * 8 + j * 256);
  };
  // stage V tile: 128 rows of 64B from Vt (row stride T_) into Vsh[buf] ([e][key])
  auto stageV = [&](int buf, int key0) {
    const _Float16* g = VtBH + key0;
    _Float16* dst = &Vsh[buf][0];
#pragma unroll
    for (int j = 0; j < 16; ++j) {
      int c   = j * 32 + lane;     // chunk id 0..511
      int row = c >> 2;            // 0..127  (e)
      int off = (c & 3) * 8;       // halfs: 0,8,16,24
      cp16(g + (long)row * T_ + off, dst + row * 32 + off);
    }
  };

  stageK(0, 0);
  stageV(0, 0);

  for (int kb = 0; kb < nkb; ++kb) {
    const int cur  = kb & 1;
    const int key0 = kb * 32;

    wait_async0();
    __syncthreads();
    if (kb + 1 < nkb) { stageK(cur ^ 1, key0 + 32); stageV(cur ^ 1, key0 + 32); }

    // S = Q K^T from LDS (two 16x16 N-tiles)
    const _Float16* kt = &Ksh[cur][0];
    v8f s0 = zero, s1 = zero;
#pragma unroll
    for (int d = 0; d < 4; ++d) {
      v16h b0 = ldBt(kt + d * 32, DH_);             // keys key0..key0+15
      v16h b1 = ldBt(kt + 16 * DH_ + d * 32, DH_);  // keys key0+16..key0+31
      s0 = wmma_f16(qa[d], b0, s0);
      s1 = wmma_f16(qa[d], b1, s1);
    }

    // scale + causal mask (branchless, wave-uniform control flow)
#pragma unroll
    for (int j = 0; j < 8; ++j) {
      int qrow = q0 + j + mrow;
      float v0 = s0[j] * scale;
      float v1 = s1[j] * scale;
      if (key0 + n > qrow)      v0 = NEG_HUGE;
      if (key0 + 16 + n > qrow) v1 = NEG_HUGE;
      s0[j] = v0; s1[j] = v1;
    }

    // online softmax: each C-row lives in one 16-lane half -> xor-reduce 1,2,4,8
    float alpha[8];
#pragma unroll
    for (int j = 0; j < 8; ++j) {
      float mx = fmaxf(s0[j], s1[j]);
#pragma unroll
      for (int off = 1; off < 16; off <<= 1) mx = fmaxf(mx, __shfl_xor(mx, off, 32));
      float mn = fmaxf(m[j], mx);
      alpha[j] = __expf(m[j] - mn);
      float p0 = __expf(s0[j] - mn);
      float p1 = __expf(s1[j] - mn);
      float rs = p0 + p1;
#pragma unroll
      for (int off = 1; off < 16; off <<= 1) rs += __shfl_xor(rs, off, 32);
      l[j] = l[j] * alpha[j] + rs;
      m[j] = mn;
      s0[j] = p0; s1[j] = p1;
    }

    // C-layout -> LDS (row-major 16x32), then reload in WMMA A layout
    __syncthreads();
#pragma unroll
    for (int j = 0; j < 8; ++j) {
      Psh[j + mrow][n]      = (_Float16)s0[j];
      Psh[j + mrow][16 + n] = (_Float16)s1[j];
    }
    __syncthreads();
    v16h pa;
    {
      const int mm  = lane & 15;
      const int kb2 = (lane < 16) ? 0 : 16;
      v8h lo = *(const v8h*)(&Psh[mm][kb2]);
      v8h hi = *(const v8h*)(&Psh[mm][kb2 + 8]);
#pragma unroll
      for (int i = 0; i < 8; ++i) { pa[i] = lo[i]; pa[i + 8] = hi[i]; }
    }

    // O = O*alpha + P V  (V B-operands contiguous from Vsh [e][key])
#pragma unroll
    for (int t = 0; t < 8; ++t)
#pragma unroll
      for (int j = 0; j < 8; ++j) o[t][j] *= alpha[j];

    const _Float16* vt = &Vsh[cur][0];
#pragma unroll
    for (int g = 0; g < 2; ++g) {
      v16h bm[4];
#pragma unroll
      for (int t = 0; t < 4; ++t)
        bm[t] = ldBt(vt + (g * 4 + t) * 16 * 32, 32);
#pragma unroll
      for (int t = 0; t < 4; ++t)
        o[g * 4 + t] = wmma_f16(pa, bm[t], o[g * 4 + t]);
    }

    __syncthreads();  // finish LDS reads before this buffer is restaged
  }

  // epilogue: O[b, t, h*Dh + e] f16 (heads concatenated)
#pragma unroll
  for (int t = 0; t < 8; ++t) {
#pragma unroll
    for (int j = 0; j < 8; ++j) {
      float val = o[t][j] / l[j];
      int trow = q0 + j + mrow;
      O[((long)b * T_ + trow) * (H_ * DH_) + h * DH_ + t * 16 + n] = (_Float16)val;
    }
  }
}

// ---------------- output projection: one wave computes 32 x 128 ----------------
// grid: (BT/32, D/128), block: 32
__global__ void __launch_bounds__(32, 1)
k_proj(const _Float16* __restrict__ O, const _Float16* __restrict__ WpT,
       const float* __restrict__ bp, float* __restrict__ out) {
  const int mb = blockIdx.x;
  const int nb = blockIdx.y;
  const int row0 = mb * 32;
  const int n0   = nb * 128;
  const int lane = threadIdx.x & 31;
  const int n    = lane & 15;
  const int mrow = (lane >= 16) ? 8 : 0;

  v8f zero = {0.f,0.f,0.f,0.f,0.f,0.f,0.f,0.f};
  v8f acc[2][8];
#pragma unroll
  for (int mt = 0; mt < 2; ++mt)
#pragma unroll
    for (int i = 0; i < 8; ++i) acc[mt][i] = zero;

  for (int k0 = 0; k0 < D_; k0 += 32) {
    v16h a0 = ldA16x32(O + (long)row0 * D_ + k0, D_);
    v16h a1 = ldA16x32(O + (long)(row0 + 16) * D_ + k0, D_);
#pragma unroll
    for (int g = 0; g < 2; ++g) {
      v16h bm[4];
#pragma unroll
      for (int t = 0; t < 4; ++t)
        bm[t] = ldBt(WpT + (long)(n0 + (g * 4 + t) * 16) * D_ + k0, D_);
#pragma unroll
      for (int t = 0; t < 4; ++t) {
        acc[0][g * 4 + t] = wmma_f16(a0, bm[t], acc[0][g * 4 + t]);
        acc[1][g * 4 + t] = wmma_f16(a1, bm[t], acc[1][g * 4 + t]);
      }
    }
  }

#pragma unroll
  for (int mt = 0; mt < 2; ++mt) {
#pragma unroll
    for (int nt = 0; nt < 8; ++nt) {
#pragma unroll
      for (int j = 0; j < 8; ++j) {
        int col = n0 + nt * 16 + n;
        out[(long)(row0 + mt * 16 + j + mrow) * D_ + col] = acc[mt][nt][j] + bp[col];
      }
    }
  }
}

// ---------------- host launch ----------------
extern "C" void kernel_launch(void* const* d_in, const int* in_sizes, int n_in,
                              void* d_out, int out_size, void* d_ws, size_t ws_size,
                              hipStream_t stream) {
  const float* x  = (const float*)d_in[0];
  const float* Wq = (const float*)d_in[1];
  const float* bq = (const float*)d_in[2];
  const float* Wk = (const float*)d_in[3];
  const float* bk = (const float*)d_in[4];
  const float* Wv = (const float*)d_in[5];
  const float* bv = (const float*)d_in[6];
  const float* Wp = (const float*)d_in[7];
  const float* bp = (const float*)d_in[8];
  float* out = (float*)d_out;

  // workspace carve-out (total ~118 MB)
  char* ws = (char*)d_ws;
  size_t off = 0;
  auto carve = [&](size_t bytes) {
    void* p = ws + off;
    off += (bytes + 255) & ~(size_t)255;
    return p;
  };
  _Float16* xh   = (_Float16*)carve((size_t)BT_ * D_ * 2);           // 16.8 MB
  _Float16* WqT  = (_Float16*)carve((size_t)H_ * D_ * DH_ * 2);      // 8.4 MB
  _Float16* WkT  = (_Float16*)carve((size_t)H_ * D_ * DH_ * 2);
  _Float16* WvT  = (_Float16*)carve((size_t)H_ * D_ * DH_ * 2);
  _Float16* WpT  = (_Float16*)carve((size_t)D_ * D_ * 2);            // 8.4 MB
  _Float16* Qb   = (_Float16*)carve((size_t)B_ * H_ * T_ * DH_ * 2); // 16.8 MB
  _Float16* Kb   = (_Float16*)carve((size_t)B_ * H_ * T_ * DH_ * 2);
  _Float16* Vtb  = (_Float16*)carve((size_t)B_ * H_ * T_ * DH_ * 2); // transposed V
  _Float16* Ob   = (_Float16*)carve((size_t)BT_ * D_ * 2);           // 16.8 MB

  // 1) conversions / transposes
  {
    int n = BT_ * D_;
    k_cvt_f32_f16<<<(n + 255) / 256, 256, 0, stream>>>(x, xh, n);
    int nw = H_ * D_ * DH_;
    k_cvt_whdt<<<(nw + 255) / 256, 256, 0, stream>>>(Wq, WqT);
    k_cvt_whdt<<<(nw + 255) / 256, 256, 0, stream>>>(Wk, WkT);
    k_cvt_whdt<<<(nw + 255) / 256, 256, 0, stream>>>(Wv, WvT);
    int np = D_ * D_;
    k_cvt_wpt<<<(np + 255) / 256, 256, 0, stream>>>(Wp, WpT);
  }

  // 2) QKV projections (WMMA, 32x128 per wave; V written transposed)
  k_qkv<<<dim3(BT_ / 32, 3 * H_), 32, 0, stream>>>(xh, WqT, WkT, WvT, bq, bk, bv, Qb, Kb, Vtb);

  // 3) causal flash attention (WMMA; async double-buffered K/V tiles in LDS)
  k_flash<<<dim3(T_ / 16, B_ * H_), 32, 0, stream>>>(Qb, Kb, Vtb, Ob);

  // 4) output projection (WMMA, 32x128 per wave) -> f32 out
  k_proj<<<dim3(BT_ / 32, D_ / 128), 32, 0, stream>>>(Ob, WpT, bp, out);
}
